// LSTMLayer_18743237280268
// MI455X (gfx1250) — matrix operand order repaired and checked by reading, not compiled, not run on hardware
//
#include <hip/hip_runtime.h>
#include <hip/hip_bf16.h>

// ---------------------------------------------------------------------------
// LSTM layer for MI455X (gfx1250, wave32, WMMA).
//   B=64, T=1024, D=256, H=512, 4H=2048.
// Strategy: persistent scan kernel; bf16 WMMA (f32 accum) on the fused
// [h_t | x_t] (K=768) GEMM; c-state in LDS; h double-buffered in global bf16;
// device-wide per-step sync via atomic counter.
// ---------------------------------------------------------------------------

#define B_      64
#define T_      1024
#define D_      256
#define H_      512
#define G4      (4 * H_)      // 2048 gate columns
#define KTOT    (H_ + D_)     // 768: fused K = hidden + input
#define NWG     32            // one WG per 16-column hidden slice
#define THREADS 512           // 16 waves: 4 gates x 4 M-tiles

typedef __bf16 bf16;
typedef __attribute__((ext_vector_type(16))) __bf16 v16bf;
typedef __attribute__((ext_vector_type(8)))  float  v8f;
typedef __attribute__((ext_vector_type(4)))  unsigned int u32x4;

union Frag { u32x4 q[2]; v16bf v; };   // two 16B loads -> one WMMA operand

// --- prep: pack [W_h ; w_x] transposed to [N=2048][K=768] bf16 --------------
__global__ void pack_weights(const float* __restrict__ Wh,
                             const float* __restrict__ wx,
                             bf16* __restrict__ Wt) {
  int id = blockIdx.x * 256 + threadIdx.x;           // id = n*KTOT + k
  if (id >= G4 * KTOT) return;
  int n = id / KTOT, k = id % KTOT;
  float v = (k < H_) ? Wh[(long)k * G4 + n] : wx[(long)(k - H_) * G4 + n];
  Wt[id] = (bf16)v;
}

// --- prep: x [B][T][D] f32 -> xT [T][B][D] bf16 -----------------------------
__global__ void pack_x(const float* __restrict__ x, bf16* __restrict__ xT) {
  int id = blockIdx.x * 256 + threadIdx.x;           // id = (t*B + b)*D + d
  if (id >= B_ * T_ * D_) return;
  int d  = id % D_;
  int tb = id / D_;
  int b  = tb % B_;
  int t  = tb / B_;
  xT[id] = (bf16)x[((long)b * T_ + t) * D_ + d];
}

// --- persistent scan --------------------------------------------------------
__global__ __launch_bounds__(THREADS, 1)
void lstm_scan(const bf16* __restrict__ Wt,    // [2048][768] bf16
               const bf16* __restrict__ xT,    // [T][B][D]  bf16
               const float* __restrict__ bias, // [2048] f32
               bf16* __restrict__ hbuf,        // [2][B][H] bf16, double buffer
               unsigned* __restrict__ cnt,     // global step counter
               float* __restrict__ out) {      // hidden_seq | h_n | c_n
  __shared__ float ldsG[4][B_][16];   // per-step gate tiles (f,i,g,o)  16KB
  __shared__ float ldsC[B_][16];      // persistent cell state slice     4KB

  const int wg   = blockIdx.x;        // hidden-column slice [16*wg, 16*wg+16)
  const int tid  = threadIdx.x;
  const int wave = tid >> 5;
  const int lane = tid & 31;
  const int g    = wave >> 2;         // gate 0..3  (f,i,g,o)
  const int mt   = wave & 3;          // M tile 0..3 (rows of batch)
  const int l16  = lane & 15;
  const int hi   = lane >> 4;         // half-wave select
  const int mrow = mt * 16 + l16;                 // batch row for A fragment
  const int ncol = g * H_ + wg * 16 + l16;        // gate column for B fragment
  const int kAoff = hi * 8;           // A layout: lanes16-31 hold K+8 / K+24
  const int kBoff = hi * 8;           // B layout: lanes16-31 hold K+8 / K+24

  const bf16* Brow = Wt + (long)ncol * KTOT;

  for (int i = tid; i < B_ * 16; i += THREADS) ldsC[i >> 4][i & 15] = 0.f;
  __syncthreads();

  for (int t = 0; t < T_; ++t) {
    const bf16* Ah = hbuf + (size_t)(t & 1) * (B_ * H_) + (long)mrow * H_;
    const bf16* Ax = xT + ((long)t * B_ + mrow) * D_;
    if (t + 1 < T_)  // warm L2/GL1 for next step's activations
      __builtin_prefetch(xT + ((long)(t + 1) * B_ + mrow) * D_, 0, 1);

    v8f acc = {0.f, 0.f, 0.f, 0.f, 0.f, 0.f, 0.f, 0.f};
#pragma unroll
    for (int kc = 0; kc < KTOT / 32; ++kc) {       // 24 chunks: 16 h + 8 x
      const int kb = kc * 32;
      const bf16* ap = (kc < H_ / 32) ? (Ah + kb + kAoff)
                                      : (Ax + (kb - H_) + kAoff);
      Frag a, b;
      a.q[0] = *(const u32x4*)(ap);
      a.q[1] = *(const u32x4*)(ap + 16);
      const bf16* bp = Brow + kb + kBoff;
      b.q[0] = *(const u32x4*)(bp);
      b.q[1] = *(const u32x4*)(bp + 16);
      acc = __builtin_amdgcn_wmma_f32_16x16x32_bf16(
          false, a.v, false, b.v, (short)0, acc, false, false);
    }

    const float bv = bias[ncol];
    {   // scatter C/D layout (row = v + 8*hi + 16*mt, col = lane&15) to LDS
      const int rbase = mt * 16 + (hi << 3);
#pragma unroll
      for (int i = 0; i < 8; ++i) ldsG[g][rbase + i][l16] = acc[i] + bv;
    }
    __syncthreads();

    // LSTM cell: 1024 (b,j) cells, 2 per thread
#pragma unroll
    for (int rep = 0; rep < 2; ++rep) {
      const int idx = tid + rep * THREADS;
      const int bb = idx >> 4, j = idx & 15;
      const float fg = ldsG[0][bb][j], ig = ldsG[1][bb][j];
      const float gg = ldsG[2][bb][j], og = ldsG[3][bb][j];
      const float f_t = 1.f / (1.f + __expf(-fg));
      const float i_t = 1.f / (1.f + __expf(-ig));
      const float g_t = tanhf(gg);
      const float o_t = 1.f / (1.f + __expf(-og));
      const float c_t = f_t * ldsC[bb][j] + i_t * g_t;
      ldsC[bb][j] = c_t;
      const float h_t = o_t * tanhf(c_t);
      const int col = wg * 16 + j;
      out[((long)bb * T_ + t) * H_ + col] = h_t;
      hbuf[(size_t)((t + 1) & 1) * (B_ * H_) + bb * H_ + col] = (bf16)h_t;
      if (t == T_ - 1) {                 // append (h_n, c_n)
        const long base = (long)B_ * T_ * H_;
        out[base + (long)bb * H_ + col]                  = h_t;
        out[base + (long)B_ * H_ + (long)bb * H_ + col]  = c_t;
      }
    }
    __syncthreads();

    if (t < T_ - 1) {                    // device-wide step barrier
      __threadfence();
      if (tid == 0) {
        __hip_atomic_fetch_add(cnt, 1u, __ATOMIC_ACQ_REL,
                               __HIP_MEMORY_SCOPE_AGENT);
        const unsigned target = (unsigned)(t + 1) * NWG;
        while (__hip_atomic_load(cnt, __ATOMIC_ACQUIRE,
                                 __HIP_MEMORY_SCOPE_AGENT) < target)
          __builtin_amdgcn_s_sleep(1);
      }
      __syncthreads();
    }
  }
}

// ---------------------------------------------------------------------------
extern "C" void kernel_launch(void* const* d_in, const int* in_sizes, int n_in,
                              void* d_out, int out_size, void* d_ws, size_t ws_size,
                              hipStream_t stream) {
  const float* x    = (const float*)d_in[0];  // [B,T,D]
  const float* W_h  = (const float*)d_in[1];  // [H,4H]
  const float* w_x  = (const float*)d_in[2];  // [D,4H]
  const float* bias = (const float*)d_in[3];  // [4H]

  char* ws = (char*)d_ws;
  const size_t WT_BYTES = (size_t)G4 * KTOT * sizeof(bf16);        // 3 MB
  const size_t XT_BYTES = (size_t)B_ * T_ * D_ * sizeof(bf16);     // 32 MB
  const size_t HB_BYTES = (size_t)2 * B_ * H_ * sizeof(bf16);      // 128 KB
  bf16*     Wt   = (bf16*)(ws);
  bf16*     xT   = (bf16*)(ws + WT_BYTES);
  bf16*     hbuf = (bf16*)(ws + WT_BYTES + XT_BYTES);
  unsigned* cnt  = (unsigned*)(ws + WT_BYTES + XT_BYTES + HB_BYTES);

  // zero h0 double-buffer + step counter (fresh every call: deterministic)
  hipMemsetAsync(hbuf, 0, HB_BYTES + sizeof(unsigned), stream);

  pack_weights<<<(G4 * KTOT + 255) / 256, 256, 0, stream>>>(W_h, w_x, Wt);
  pack_x<<<(B_ * T_ * D_ + 255) / 256, 256, 0, stream>>>(x, xT);
  lstm_scan<<<NWG, THREADS, 0, stream>>>(Wt, xT, bias, hbuf, cnt,
                                         (float*)d_out);
}
